// SharedConvs_44822278701235
// MI455X (gfx1250) — matrix operand order repaired and checked by reading, not compiled, not run on hardware
//
#include <hip/hip_runtime.h>

typedef __attribute__((ext_vector_type(16))) _Float16 v16h;
typedef __attribute__((ext_vector_type(8)))  float    v8f;

#define ICH 32
#define SEGW 128                    // output columns per block
#define STG_PX (SEGW + 2)           // staged pixels per row (1px halo each side)
#define ROW_HALFS (STG_PX * ICH)    // 4160 halfs per staged row
#define NWAVES 8
#define BLOCK (NWAVES * 32)
#define W_HALFS (9 * 2 * 2 * 16 * 16)   // 9216 repacked weight halfs (18 KB)
#define STRIP 8                     // output rows per block (non-pool)
#define STRIPP 4                    // output row-pairs per block (pool)
#define NROWS_STG 10                // staged input rows (both variants)

// ---------------------------------------------------------------------------
// Repack w (32 och x 288 k, f32, k = (kh*3+kw)*32 + c) into f16 B-fragment
// order: [t(9)][nt(2)][h(2)][N(16)][j(16)], where for tap t, channel-tile nt,
// lane (N + 16*h) of the wave holds B elements K = t*32 + h*16 + j (j=0..15),
// column N + 16*nt. Each lane then loads 16 contiguous halfs (2 x b128).
// ---------------------------------------------------------------------------
__global__ void repack_w_kernel(const float* __restrict__ w,
                                _Float16* __restrict__ wB) {
  int idx = blockIdx.x * blockDim.x + threadIdx.x;
  if (idx >= W_HALFS) return;
  int j  = idx & 15;
  int N  = (idx >> 4) & 15;
  int h  = (idx >> 8) & 1;
  int nt = (idx >> 9) & 1;
  int t  = idx >> 10;
  int och = nt * 16 + N;
  int k   = t * 32 + h * 16 + j;            // (kh,kw) tap t, channel h*16+j
  wB[idx] = (_Float16)w[och * 288 + k];
}

// ---------------------------------------------------------------------------
// 3x3 SAME conv, 32->32 ch, + bias + ReLU, optional fused 2x2 maxpool.
// IN_F32:  input activations are f32 (layer 1), else f16.
// POOL:    process row-pairs and write 2x2-maxpooled output (H/2 x W/2).
// OUT_F32: write f32 (final layer -> d_out), else f16.
// Block = 8 waves; each wave owns 16 output columns; block owns SEGW columns
// of an 8-row strip (pool: 4 row-pairs). 10 input rows staged per tile gives
// ~1.25x input reads instead of 3x. All 18 B fragments (144 VGPRs) are loaded
// once per kernel and kept in registers; __launch_bounds__(256,2) raises the
// VGPR budget so they are never spilled.
// ---------------------------------------------------------------------------
template <bool IN_F32, bool POOL, bool OUT_F32>
__global__ void __launch_bounds__(BLOCK, 2)
conv3x3_wmma_kernel(const void* __restrict__ in_,
                    const _Float16* __restrict__ wB,
                    const float* __restrict__ bias,
                    void* __restrict__ out_,
                    int N, int H, int W, int numTiles) {
  __shared__ _Float16 smem[NROWS_STG * ROW_HALFS];   // 10 staged rows, ~83 KB

  const int lane = threadIdx.x & 31;
  const int m    = lane & 15;                // row-in-tile / column-in-tile id
  const int h    = lane >> 4;                // lane half
  const int wv   = threadIdx.x >> 5;
  const int x0   = wv * 16;                  // this wave's column offset in seg

  // ---- load all 18 B fragments into VGPRs, kept for the whole kernel ----
  v16h Bf[9][2];
  {
    const uint4* wB4 = (const uint4*)wB;
#pragma unroll
    for (int t = 0; t < 9; ++t)
#pragma unroll
      for (int nt = 0; nt < 2; ++nt) {
        int base = (((t * 2 + nt) * 2 + h) * 16 + m) * 2;  // uint4 units
        union { uint4 u[2]; v16h v; } tmp;
        tmp.u[0] = wB4[base];
        tmp.u[1] = wB4[base + 1];
        Bf[t][nt] = tmp.v;
      }
  }
  const float bv0 = bias[m];
  const float bv1 = bias[m + 16];

  const int nSegs = W / SEGW;
  const int nRowT = POOL ? (H >> 1) / STRIPP : H / STRIP;
  const int chunksPerRow = ROW_HALFS / 8;    // uint4 (8-half) chunks = 520

  for (int tile = blockIdx.x; tile < numTiles; tile += gridDim.x) {
    int seg = tile % nSegs;
    int rt  = (tile / nSegs) % nRowT;
    int n   = tile / (nSegs * nRowT);
    int y0  = POOL ? (rt * (2 * STRIPP)) : (rt * STRIP);  // first output row
    int segBase = seg * SEGW;

    __syncthreads();  // previous iteration's reads of smem are done

    // ---- stage input rows y0-1 .. y0+8 into LDS as f16 ----
    for (int c = threadIdx.x; c < NROWS_STG * chunksPerRow; c += BLOCK) {
      int r   = c / chunksPerRow;
      int cc  = c % chunksPerRow;
      int p   = cc >> 2;                     // staged pixel (0..129)
      int sub = cc & 3;                      // 8-channel chunk within pixel
      int gy  = y0 - 1 + r;
      int gx  = segBase - 1 + p;
      union { uint4 u; _Float16 f[8]; } val;
      if (gy >= 0 && gy < H && gx >= 0 && gx < W) {
        size_t pixBase = ((size_t)n * H + gy) * W + gx;
        if (IN_F32) {
          const float4* src = (const float4*)in_;
          float4 f0 = src[pixBase * 8 + sub * 2];
          float4 f1 = src[pixBase * 8 + sub * 2 + 1];
          val.f[0] = (_Float16)f0.x;  val.f[1] = (_Float16)f0.y;
          val.f[2] = (_Float16)f0.z;  val.f[3] = (_Float16)f0.w;
          val.f[4] = (_Float16)f1.x;  val.f[5] = (_Float16)f1.y;
          val.f[6] = (_Float16)f1.z;  val.f[7] = (_Float16)f1.w;
        } else {
          const uint4* src = (const uint4*)in_;
          val.u = src[pixBase * 4 + sub];
        }
      } else {
        val.u.x = 0u; val.u.y = 0u; val.u.z = 0u; val.u.w = 0u;
      }
      ((uint4*)smem)[c] = val.u;             // c == r*chunksPerRow + cc
    }
    __syncthreads();

    const uint4* sm4 = (const uint4*)smem;

    // ---- compute the strip: per row(-pair), 9 taps x 2 ch-tiles WMMAs ----
    const int nIters = POOL ? STRIPP : STRIP;
    for (int rr = 0; rr < nIters; ++rr) {
      int rbase = POOL ? (2 * rr) : rr;      // first staged row for kh=0
      v8f acc00 = {}, acc01 = {}, acc10 = {}, acc11 = {};
#pragma unroll
      for (int t = 0; t < 9; ++t) {
        const int kh = t / 3, kw = t % 3;
        int pix = x0 + m + kw;               // staged pixel for A row m
        int a0 = ((rbase + kh) * ROW_HALFS + pix * ICH + h * 16) >> 3;
        union { uint4 u[2]; v16h v; } A;
        A.u[0] = sm4[a0];
        A.u[1] = sm4[a0 + 1];
        acc00 = __builtin_amdgcn_wmma_f32_16x16x32_f16(false, A.v, false, Bf[t][0],
                                                       (short)0, acc00, false, false);
        acc01 = __builtin_amdgcn_wmma_f32_16x16x32_f16(false, A.v, false, Bf[t][1],
                                                       (short)0, acc01, false, false);
        if (POOL) {                          // second row of the pair
          int a1 = a0 + (ROW_HALFS >> 3);
          union { uint4 u[2]; v16h v; } A2;
          A2.u[0] = sm4[a1];
          A2.u[1] = sm4[a1 + 1];
          acc10 = __builtin_amdgcn_wmma_f32_16x16x32_f16(false, A2.v, false, Bf[t][0],
                                                         (short)0, acc10, false, false);
          acc11 = __builtin_amdgcn_wmma_f32_16x16x32_f16(false, A2.v, false, Bf[t][1],
                                                         (short)0, acc11, false, false);
        }
      }

      // ---- epilogue: bias + ReLU (+ 2x2 maxpool), store ----
      if (!POOL) {
        int y = y0 + rr;
        size_t rowBase = ((size_t)n * H + y) * W;
#pragma unroll
        for (int i = 0; i < 8; ++i) {
          int px = segBase + x0 + h * 8 + i; // D: vgpr i -> pixel (h*8 + i)
          float v0 = fmaxf(acc00[i] + bv0, 0.0f);
          float v1 = fmaxf(acc01[i] + bv1, 0.0f);
          size_t o = (rowBase + px) * ICH;
          if (OUT_F32) {
            float* out = (float*)out_;
            out[o + m] = v0;  out[o + m + 16] = v1;
          } else {
            _Float16* out = (_Float16*)out_;
            out[o + m] = (_Float16)v0;  out[o + m + 16] = (_Float16)v1;
          }
        }
      } else {
        int Ho = H >> 1, Wo = W >> 1;
        int oy = rt * STRIPP + rr;
        size_t rowBase = ((size_t)n * Ho + oy) * Wo;
#pragma unroll
        for (int i2 = 0; i2 < 4; ++i2) {     // pool pixel pairs (2*i2, 2*i2+1)
          float p0 = fmaxf(fmaxf(acc00[2 * i2], acc00[2 * i2 + 1]),
                           fmaxf(acc10[2 * i2], acc10[2 * i2 + 1]));
          float p1 = fmaxf(fmaxf(acc01[2 * i2], acc01[2 * i2 + 1]),
                           fmaxf(acc11[2 * i2], acc11[2 * i2 + 1]));
          p0 = fmaxf(p0 + bv0, 0.0f);        // bias uniform over the window
          p1 = fmaxf(p1 + bv1, 0.0f);
          int opx = ((segBase + x0 + h * 8) >> 1) + i2;
          size_t o = (rowBase + opx) * ICH;
          if (OUT_F32) {
            float* out = (float*)out_;
            out[o + m] = p0;  out[o + m + 16] = p1;
          } else {
            _Float16* out = (_Float16*)out_;
            out[o + m] = (_Float16)p0;  out[o + m + 16] = (_Float16)p1;
          }
        }
      }
    }
  }
}

extern "C" void kernel_launch(void* const* d_in, const int* in_sizes, int n_in,
                              void* d_out, int out_size, void* d_ws, size_t ws_size,
                              hipStream_t stream) {
  (void)in_sizes; (void)n_in; (void)out_size; (void)ws_size;
  const float* x = (const float*)d_in[0];   // (16,256,256,32) f32
  const float* w = (const float*)d_in[1];   // (32,288) f32
  const float* b = (const float*)d_in[2];   // (32,) f32

  char* ws = (char*)d_ws;
  _Float16* wB = (_Float16*)ws;                                   // 18 KB
  _Float16* h1 = (_Float16*)(ws + (size_t)32768);                 // 64 MB
  _Float16* p1 = (_Float16*)(ws + (size_t)32768 + 67108864);      // 16 MB
  _Float16* h3 = (_Float16*)(ws + (size_t)32768 + 67108864 + 16777216); // 16 MB

  repack_w_kernel<<<(W_HALFS + 255) / 256, 256, 0, stream>>>(w, wB);

  const int N = 16;
  // L1: conv+relu @256x256, f32 in -> f16 out
  {
    int H = 256, W = 256;
    int tiles = N * (H / STRIP) * (W / SEGW);       // 1024
    conv3x3_wmma_kernel<true, false, false><<<tiles, BLOCK, 0, stream>>>(
        x, wB, b, h1, N, H, W, tiles);
  }
  // L2: conv+relu+pool @256x256 -> 128x128, f16 -> f16
  {
    int H = 256, W = 256;
    int tiles = N * ((H / 2) / STRIPP) * (W / SEGW);  // 1024
    conv3x3_wmma_kernel<false, true, false><<<tiles, BLOCK, 0, stream>>>(
        h1, wB, b, p1, N, H, W, tiles);
  }
  // L3: conv+relu @128x128, f16 -> f16
  {
    int H = 128, W = 128;
    int tiles = N * (H / STRIP) * (W / SEGW);       // 256
    conv3x3_wmma_kernel<false, false, false><<<tiles, BLOCK, 0, stream>>>(
        p1, wB, b, h3, N, H, W, tiles);
  }
  // L4: conv+relu+pool @128x128 -> 64x64, f16 -> f32 (final output)
  {
    int H = 128, W = 128;
    int tiles = N * ((H / 2) / STRIPP) * (W / SEGW);  // 256
    conv3x3_wmma_kernel<false, true, true><<<tiles, BLOCK, 0, stream>>>(
        h3, wB, b, d_out, N, H, W, tiles);
  }
}
